// L2MultiHeadAttention_36258113912834
// MI455X (gfx1250) — matrix operand order repaired and checked by reading, not compiled, not run on hardware
//
#include <hip/hip_runtime.h>
#include <hip/hip_bf16.h>

#define B_  2
#define S_  2048
#define D_  1024
#define H_  16
#define DH_ 64
#define MR_ (B_ * S_)          // 4096 token rows

typedef __bf16 bf16_t;
typedef __attribute__((ext_vector_type(16))) __bf16 v16bf;
typedef __attribute__((ext_vector_type(8)))  __bf16 v8bf;
typedef __attribute__((ext_vector_type(8)))  float  v8f;

__device__ __forceinline__ bf16_t f2bf(float f) { return (bf16_t)f; }
__device__ __forceinline__ float  bf2f(bf16_t h) { return (float)h; }

__device__ __forceinline__ v8f zero8() {
  v8f z;
#pragma unroll
  for (int e = 0; e < 8; ++e) z[e] = 0.0f;
  return z;
}

__device__ __forceinline__ v8f wmma_bf16(v16bf a, v16bf b, v8f c) {
  // D = A(16x32 bf16) * B(32x16 bf16) + C(16x16 f32)
  return __builtin_amdgcn_wmma_f32_16x16x32_bf16(
      /*neg_a=*/false, a, /*neg_b=*/false, b,
      /*c_mod=*/(short)0, c, /*reuse_a=*/false, /*reuse_b=*/false);
}

// ---- fragment loaders (wave32, same contiguous k order for A and B) -------------
// A: 16x32 tile from row-major bf16 source (leading dim ld, elements).
// lane: row m = lane&15, half = lane>>4; K = half*16 + e (contiguous).
__device__ __forceinline__ v16bf load_a_frag(const bf16_t* base, int ld, int lane) {
  const int m = lane & 15, half = lane >> 4;
  const bf16_t* p = base + (size_t)m * ld + half * 16;
  v8bf lo = *(const v8bf*)(p);
  v8bf hi = *(const v8bf*)(p + 8);
  return __builtin_shufflevector(lo, hi, 0,1,2,3,4,5,6,7,8,9,10,11,12,13,14,15);
}

// B: 32x16 tile where the source stores B^T row-major, i.e. src[n][k]
// (leading dim ld). lane: col n = lane&15; K = half*16 + e (contiguous).
__device__ __forceinline__ v16bf load_b_frag_nmajor(const bf16_t* base, int ld, int lane) {
  const int n = lane & 15, half = lane >> 4;
  const bf16_t* p = base + (size_t)n * ld + half * 16;
  v8bf lo = *(const v8bf*)(p);
  v8bf hi = *(const v8bf*)(p + 8);
  return __builtin_shufflevector(lo, hi, 0,1,2,3,4,5,6,7,8,9,10,11,12,13,14,15);
}

// B: 32x16 tile from true row-major B[k][n] source (strided gather, LDS).
__device__ __forceinline__ v16bf load_b_frag_kmajor(const bf16_t* base, int ld, int lane) {
  const int n = lane & 15, half = lane >> 4;
  const bf16_t* p = base + (size_t)(half * 16) * ld + n;
  v16bf b;
#pragma unroll
  for (int e = 0; e < 16; ++e) b[e] = p[(size_t)e * ld];
  return b;
}

// ---- kernel 0: fp32 -> bf16 elementwise (4 elems/thread) -----------------
__global__ void cvt4_kernel(const float* __restrict__ in, bf16_t* __restrict__ out, int n4) {
  int i = (blockIdx.x * blockDim.x + threadIdx.x);
  if (i >= n4) return;
  float4 v = ((const float4*)in)[i];
  bf16_t* o = out + (size_t)i * 4;
  o[0] = f2bf(v.x); o[1] = f2bf(v.y); o[2] = f2bf(v.z); o[3] = f2bf(v.w);
}

// ---- kernel 1/5: C[32x64 per wave] = A * Bw^T (+bias)(+resid) ------------
// A: [M x K] bf16 row-major.  Bw: [N x K] bf16 row-major (i.e. B^T).
// 8 WMMAs per k-step against 6 fragment loads (2 A + 4 B) for better AI.
template <bool BF16_OUT, bool RESIDUAL>
__global__ __launch_bounds__(32)
void gemm32x64_nt(const bf16_t* __restrict__ A, const bf16_t* __restrict__ Bw,
                  const float* __restrict__ bias, const float* __restrict__ resid,
                  void* __restrict__ Cout, int M, int N, int K) {
  const int lane  = threadIdx.x;
  const int mbase = blockIdx.x * 32;
  const int nbase = blockIdx.y * 64;
  v8f c[2][4];
#pragma unroll
  for (int mt = 0; mt < 2; ++mt)
#pragma unroll
    for (int t = 0; t < 4; ++t) c[mt][t] = zero8();

  for (int kk = 0; kk < K; kk += 32) {
    v16bf a0 = load_a_frag(A + (size_t)mbase * K + kk, K, lane);
    v16bf a1 = load_a_frag(A + (size_t)(mbase + 16) * K + kk, K, lane);
#pragma unroll
    for (int t = 0; t < 4; ++t) {
      v16bf b = load_b_frag_nmajor(Bw + (size_t)(nbase + t * 16) * K + kk, K, lane);
      c[0][t] = wmma_bf16(a0, b, c[0][t]);
      c[1][t] = wmma_bf16(a1, b, c[1][t]);
    }
  }

  const int n = lane & 15, half = lane >> 4;
#pragma unroll
  for (int mt = 0; mt < 2; ++mt) {
#pragma unroll
    for (int t = 0; t < 4; ++t) {
#pragma unroll
      for (int r = 0; r < 8; ++r) {
        const int row = mbase + mt * 16 + r + 8 * half;
        const int col = nbase + t * 16 + n;
        float v = c[mt][t][r] + bias[col];
        if (RESIDUAL) v += resid[(size_t)row * N + col];
        if (BF16_OUT) ((bf16_t*)Cout)[(size_t)row * N + col] = f2bf(v);
        else          ((float*)Cout)[(size_t)row * N + col]  = v;
      }
    }
  }
}

// ---- kernel 2: sq[b][h][s] = sum_j q[b,s,h*64+j]^2 -----------------------
__global__ void sq_kernel(const bf16_t* __restrict__ q, float* __restrict__ sq) {
  int t = blockIdx.x * blockDim.x + threadIdx.x;   // 0 .. B*H*S-1, layout [b][h][s]
  if (t >= B_ * H_ * S_) return;
  int s = t % S_;
  int bh = t / S_;
  int b = bh / H_, h = bh % H_;
  const bf16_t* p = q + (size_t)(b * S_ + s) * D_ + h * DH_;
  float acc = 0.0f;
#pragma unroll 8
  for (int j = 0; j < DH_; ++j) { float v = bf2f(p[j]); acc += v * v; }
  sq[t] = acc;
}

// ---- kernel 3: Mt[h][f][e] = (sum_i Wq[i][h*64+e] * Wv[i][h*64+f]) / 8 ---
__global__ void headmat_kernel(const float* __restrict__ Wq, const float* __restrict__ Wv,
                               bf16_t* __restrict__ Mt) {
  int h   = blockIdx.x;
  int idx = blockIdx.y * blockDim.x + threadIdx.x;   // 0..4095
  int f = idx >> 6, e = idx & 63;
  const float* wq = Wq + h * DH_ + e;
  const float* wv = Wv + h * DH_ + f;
  float acc = 0.0f;
  for (int i = 0; i < D_; ++i)
    acc += wq[(size_t)i * D_] * wv[(size_t)i * D_];
  Mt[(size_t)h * DH_ * DH_ + f * DH_ + e] = f2bf(acc * 0.125f);
}

// ---- kernel 4: flash-style L2-distance attention + M_h fusion ------------
// 4 waves per block share each staged 32x64 Q_j tile (4x less global q
// traffic); each wave owns a distinct 16-row query tile.
#define QLD 80   // padded LDS stride for 32x64 Q_j tile (bank-conflict relief)
#define OLD 72   // padded LDS stride for 16x64 O tile
#define WPB 4    // waves per block

__global__ __launch_bounds__(32 * WPB)
void attn_kernel(const bf16_t* __restrict__ q, const float* __restrict__ sq,
                 const int* __restrict__ mask, const bf16_t* __restrict__ Mt,
                 bf16_t* __restrict__ attn2) {
  __shared__ __align__(16) bf16_t q_lds[32 * QLD];
  __shared__ __align__(16) bf16_t p_lds_all[WPB][16 * 32];
  __shared__ __align__(16) bf16_t o_lds_all[WPB][16 * OLD];

  const int tid   = threadIdx.x;
  const int lane  = tid & 31;
  const int wave  = tid >> 5;
  const int bh    = blockIdx.x;            // 0..31
  const int b     = bh / H_, h = bh % H_;
  const int irow  = (blockIdx.y * WPB + wave) * 16;  // query row tile
  const int n     = lane & 15, half = lane >> 4;

  bf16_t* p_lds = p_lds_all[wave];
  bf16_t* o_lds = o_lds_all[wave];

  const bf16_t* qh  = q + (size_t)b * S_ * D_ + h * DH_;   // row stride D_
  const float*  sqh = sq + ((size_t)b * H_ + h) * S_;

  // Q_i A-fragments (16 rows x 64 K)
  v16bf ai0 = load_a_frag(qh + (size_t)irow * D_,      D_, lane);
  v16bf ai1 = load_a_frag(qh + (size_t)irow * D_ + 32, D_, lane);

  float sqi[8];
#pragma unroll
  for (int r = 0; r < 8; ++r) sqi[r] = sqh[irow + r + 8 * half];

  float mstate[8], lstate[8];
#pragma unroll
  for (int r = 0; r < 8; ++r) { mstate[r] = -3.0e30f; lstate[r] = 0.0f; }
  v8f o[4];
#pragma unroll
  for (int t = 0; t < 4; ++t) o[t] = zero8();

  const float NEG_BIG = -9.0e15f;

  for (int j = 0; j < S_; j += 32) {
    // cooperative stage of Q_j (32 rows x 64 cols): 128 threads, 32B each
    {
      const int row = tid >> 2, seg = tid & 3;
      const bf16_t* src = qh + (size_t)(j + row) * D_ + seg * 16;
      bf16_t* dst = &q_lds[row * QLD + seg * 16];
      *(v8bf*)(dst)     = *(const v8bf*)(src);
      *(v8bf*)(dst + 8) = *(const v8bf*)(src + 8);
    }
    __syncthreads();

    // logits tiles: s0 = Q_i x Q_j[0:16]^T, s1 = Q_i x Q_j[16:32]^T (K = 64)
    v8f s0 = zero8(), s1 = zero8();
    s0 = wmma_bf16(ai0, load_b_frag_nmajor(q_lds,                QLD, lane), s0);
    s0 = wmma_bf16(ai1, load_b_frag_nmajor(q_lds + 32,           QLD, lane), s0);
    s1 = wmma_bf16(ai0, load_b_frag_nmajor(q_lds + 16 * QLD,     QLD, lane), s1);
    s1 = wmma_bf16(ai1, load_b_frag_nmajor(q_lds + 16 * QLD + 32,QLD, lane), s1);

    const int t0 = j + n, t1 = j + 16 + n;
    const float sq0 = sqh[t0], sq1 = sqh[t1];
    const int   mk0 = mask[b * S_ + t0], mk1 = mask[b * S_ + t1];

    float lg0[8], lg1[8], rowm[8];
#pragma unroll
    for (int r = 0; r < 8; ++r) {
      lg0[r] = mk0 ? (2.0f * s0[r] - sqi[r] - sq0) * 0.125f : NEG_BIG;
      lg1[r] = mk1 ? (2.0f * s1[r] - sqi[r] - sq1) * 0.125f : NEG_BIG;
      rowm[r] = fmaxf(lg0[r], lg1[r]);
    }
    // row-max across the 16 lanes of each half-group
#pragma unroll
    for (int off = 1; off < 16; off <<= 1)
#pragma unroll
      for (int r = 0; r < 8; ++r)
        rowm[r] = fmaxf(rowm[r], __shfl_xor(rowm[r], off, 32));

    float scale[8];
#pragma unroll
    for (int r = 0; r < 8; ++r) {
      float mn = fmaxf(mstate[r], rowm[r]);
      scale[r] = __expf(mstate[r] - mn);
      mstate[r] = mn;
      lstate[r] *= scale[r];
    }
#pragma unroll
    for (int t = 0; t < 4; ++t)
#pragma unroll
      for (int r = 0; r < 8; ++r) o[t][r] *= scale[r];

#pragma unroll
    for (int r = 0; r < 8; ++r) {
      float p0 = __expf(lg0[r] - mstate[r]);
      float p1 = __expf(lg1[r] - mstate[r]);
      lstate[r] += p0 + p1;                       // per-lane partial row sum
      p_lds[(r + 8 * half) * 32 + n]      = f2bf(p0);
      p_lds[(r + 8 * half) * 32 + 16 + n] = f2bf(p1);
    }
    __syncthreads();   // p_lds visible; q_lds reads below still pending

    // O += P(16x32) x Q_j(32x64)
    v16bf pa = load_a_frag(p_lds, 32, lane);
#pragma unroll
    for (int dt = 0; dt < 4; ++dt) {
      v16bf bv = load_b_frag_kmajor(q_lds + dt * 16, QLD, lane);
      o[dt] = wmma_bf16(pa, bv, o[dt]);
    }
    __syncthreads();   // before next iteration overwrites q_lds / p_lds
  }

  // full row sums of l across half-group lanes
#pragma unroll
  for (int off = 1; off < 16; off <<= 1)
#pragma unroll
    for (int r = 0; r < 8; ++r) lstate[r] += __shfl_xor(lstate[r], off, 32);

  // normalize O and park it in LDS (row-major 16x64, padded stride)
#pragma unroll
  for (int dt = 0; dt < 4; ++dt)
#pragma unroll
    for (int r = 0; r < 8; ++r)
      o_lds[(r + 8 * half) * OLD + dt * 16 + n] = f2bf(o[dt][r] / lstate[r]);
  __syncthreads();

  // attn2(16x64) = O_norm(16x64) x M_h(64x64); Mt stores M_h^T row-major
  v16bf oa0 = load_a_frag(o_lds,      OLD, lane);
  v16bf oa1 = load_a_frag(o_lds + 32, OLD, lane);
  const bf16_t* mth = Mt + (size_t)h * DH_ * DH_;
  v8f c2[4];
#pragma unroll
  for (int t = 0; t < 4; ++t) c2[t] = zero8();
#pragma unroll
  for (int ft = 0; ft < 4; ++ft) {
    c2[ft] = wmma_bf16(oa0, load_b_frag_nmajor(mth + (size_t)(ft * 16) * DH_,      DH_, lane), c2[ft]);
    c2[ft] = wmma_bf16(oa1, load_b_frag_nmajor(mth + (size_t)(ft * 16) * DH_ + 32, DH_, lane), c2[ft]);
  }
#pragma unroll
  for (int ft = 0; ft < 4; ++ft)
#pragma unroll
    for (int r = 0; r < 8; ++r) {
      const int row = b * S_ + irow + r + 8 * half;
      attn2[(size_t)row * D_ + h * DH_ + ft * 16 + n] = f2bf(c2[ft][r]);
    }
}

// ---- host-side orchestration ---------------------------------------------
extern "C" void kernel_launch(void* const* d_in, const int* in_sizes, int n_in,
                              void* d_out, int out_size, void* d_ws, size_t ws_size,
                              hipStream_t stream) {
  (void)in_sizes; (void)n_in; (void)out_size; (void)ws_size;
  const float* x    = (const float*)d_in[0];
  const float* Wq   = (const float*)d_in[1];
  const float* bq   = (const float*)d_in[2];
  const float* Wv   = (const float*)d_in[3];
  const float* Wout = (const float*)d_in[4];
  const float* bout = (const float*)d_in[5];
  const int*   mask = (const int*)d_in[6];

  char* ws = (char*)d_ws;
  auto carve = [&](size_t bytes) -> char* {
    char* p = ws; ws += (bytes + 255) & ~(size_t)255; return p;
  };
  bf16_t* x_bf    = (bf16_t*)carve((size_t)MR_ * D_ * 2);
  bf16_t* Wq_bf   = (bf16_t*)carve((size_t)D_ * D_ * 2);
  bf16_t* Wout_bf = (bf16_t*)carve((size_t)D_ * D_ * 2);
  bf16_t* q_bf    = (bf16_t*)carve((size_t)MR_ * D_ * 2);
  bf16_t* attn2   = (bf16_t*)carve((size_t)MR_ * D_ * 2);
  float*  sqw     = (float*)carve((size_t)B_ * H_ * S_ * 4);
  bf16_t* Mt      = (bf16_t*)carve((size_t)H_ * DH_ * DH_ * 2);

  // 0) fp32 -> bf16 conversions
  cvt4_kernel<<<dim3((MR_ * D_ / 4 + 255) / 256), 256, 0, stream>>>(x, x_bf, MR_ * D_ / 4);
  cvt4_kernel<<<dim3((D_ * D_ / 4 + 255) / 256), 256, 0, stream>>>(Wq, Wq_bf, D_ * D_ / 4);
  cvt4_kernel<<<dim3((D_ * D_ / 4 + 255) / 256), 256, 0, stream>>>(Wout, Wout_bf, D_ * D_ / 4);

  // 1) q = x @ Wq^T + bq  (bf16 out)
  gemm32x64_nt<true, false><<<dim3(MR_ / 32, D_ / 64), 32, 0, stream>>>(
      x_bf, Wq_bf, bq, nullptr, q_bf, MR_, D_, D_);

  // 2) per-(b,h,s) squared norms of q
  sq_kernel<<<dim3((B_ * H_ * S_ + 255) / 256), 256, 0, stream>>>(q_bf, sqw);

  // 3) fused head matrices Mt[h] = (Wq_T[h] @ v[h])^T / 8
  headmat_kernel<<<dim3(H_, (DH_ * DH_) / 256), 256, 0, stream>>>(Wq, Wv, Mt);

  // 4) attention: softmax(L2 logits) @ q @ M_h  -> attn2 (bf16)
  attn_kernel<<<dim3(B_ * H_, S_ / (16 * WPB)), 32 * WPB, 0, stream>>>(
      q_bf, sqw, mask, Mt, attn2);

  // 5) out = x + attn2 @ Wout^T + bout  (fp32 out)
  gemm32x64_nt<false, true><<<dim3(MR_ / 32, D_ / 64), 32, 0, stream>>>(
      attn2, Wout_bf, bout, x, d_out, MR_, D_, D_);
}
